// UTUV1MoE_20641612824696
// MI455X (gfx1250) — matrix operand order repaired and checked by reading, not compiled, not run on hardware
//
#include <hip/hip_runtime.h>

// ---------------------------------------------------------------------------
// MoE (DeepSeek-style) for gfx1250: sparse-dispatch routed experts + shared
// expert. bf16 WMMA (v_wmma_f32_16x16x32_bf16) with f32 accumulation,
// bf16 intermediates, async global->LDS staging for A tiles in the down-proj
// GEMMs (global_load_async_to_lds_b128 + s_wait_asynccnt).
// Workspace usage: ~21 MB of d_ws.
// ---------------------------------------------------------------------------

#define T_TOK 2048
#define HID   1024
#define IEXP  512
#define NEXP  64
#define ISH   1024

typedef __attribute__((ext_vector_type(16))) __bf16 v16bf;
typedef __attribute__((ext_vector_type(8)))  float  v8f;

union FragU { v16bf v; unsigned int u[8]; };

static __device__ __forceinline__ unsigned int f32_to_bf16(float f) {
  unsigned int a = __float_as_uint(f);
  return (a + 0x7FFFu + ((a >> 16) & 1u)) >> 16;       // RNE f32 -> bf16
}

static __device__ __forceinline__ unsigned int pack2_bf16(float lo, float hi) {
  return f32_to_bf16(lo) | (f32_to_bf16(hi) << 16);
}

static __device__ __forceinline__ float silu_f(float x) {
  return x / (1.0f + __expf(-x));
}

// Async copy 16 bytes global -> LDS (per lane), tracked by ASYNCcnt.
static __device__ __forceinline__ void async_g2l_b128(unsigned lds_off, const void* gaddr) {
  asm volatile("global_load_async_to_lds_b128 %0, %1, off"
               :: "v"(lds_off), "v"(gaddr) : "memory");
}
static __device__ __forceinline__ void async_wait0() {
  asm volatile("s_wait_asynccnt 0x0" ::: "memory");
}

// A-fragment, 16x32 bf16, row-major row in LDS (row = contiguous 64 halfs;
// `row` points at the lane's row, in dword units, 2 halfs per dword).
// Layout: lanes 0-15 hold K=0..15; lanes 16-31 hold K=16..31 (matches frag_b).
static __device__ __forceinline__ v16bf frag_a(const unsigned int* row, int k0, int lane) {
  int base = (k0 >> 1) + ((lane & 16) ? 8 : 0);
  FragU f;
  f.u[0] = row[base + 0]; f.u[1] = row[base + 1];
  f.u[2] = row[base + 2]; f.u[3] = row[base + 3];
  f.u[4] = row[base + 4]; f.u[5] = row[base + 5];
  f.u[6] = row[base + 6]; f.u[7] = row[base + 7];
  return f.v;
}

// B-fragment, 32x16 bf16, stored TRANSPOSED in LDS ([N][K], contiguous K).
// lanes 0-15 read K=k0..k0+15 of column N=lane; lanes 16-31 read K=k0+16..k0+31.
static __device__ __forceinline__ v16bf frag_b(const unsigned int* col, int k0, int lane) {
  int base = (k0 >> 1) + ((lane & 16) ? 8 : 0);
  FragU f;
#pragma unroll
  for (int i = 0; i < 8; ++i) f.u[i] = col[base + i];
  return f.v;
}

static __device__ __forceinline__ v8f wmma_bf16(v16bf a, v16bf b, v8f c) {
  return __builtin_amdgcn_wmma_f32_16x16x32_bf16(
      /*neg_a=*/false, a, /*neg_b=*/false, b,
      /*c_mod=*/(short)0, c, /*reuse_a=*/false, /*reuse_b=*/false);
}

// ---------------------------------------------------------------------------
// Router: one 64-thread block per token.
// ---------------------------------------------------------------------------
__global__ __launch_bounds__(64) void moe_router(
    const float* __restrict__ x, const float* __restrict__ rw,
    const float* __restrict__ ebias, float* __restrict__ coef,
    int* __restrict__ topk, int* __restrict__ counts) {
  __shared__ float xs[HID];
  __shared__ float sc[NEXP];
  __shared__ float sfc[NEXP];
  __shared__ float cw[NEXP];
  const int t = blockIdx.x;
  const int e = threadIdx.x;
  for (int i = e; i < HID; i += 64) xs[i] = x[(size_t)t * HID + i];
  cw[e] = 0.0f;
  __syncthreads();
  const float* w = rw + (size_t)e * HID;
  float acc = 0.0f;
#pragma unroll 4
  for (int i = 0; i < HID; ++i) acc += xs[i] * w[i];
  float s = 1.0f / (1.0f + __expf(-acc));
  sc[e] = s;
  sfc[e] = s + ebias[e];
  __syncthreads();
  if (e == 0) {
    float gs[8];
    for (int g = 0; g < 8; ++g) {           // group score = sum of top-2 of 8
      float m1 = -1e30f, m2 = -1e30f;
      for (int j = 0; j < 8; ++j) {
        float v = sfc[g * 8 + j];
        if (v > m1) { m2 = m1; m1 = v; } else if (v > m2) { m2 = v; }
      }
      gs[g] = m1 + m2;
    }
    unsigned gmask = 0;
    for (int k = 0; k < 4; ++k) {           // top-4 groups
      int bi = 0; float bv = -1e30f;
      for (int g = 0; g < 8; ++g)
        if (!((gmask >> g) & 1u) && gs[g] > bv) { bv = gs[g]; bi = g; }
      gmask |= 1u << bi;
    }
    unsigned long long taken = 0ull;
    int idx[8]; float wsum = 0.0f;
    for (int k = 0; k < 8; ++k) {           // top-8 experts within selected groups
      int bi = 0; float bv = -1e30f;
      for (int ee = 0; ee < NEXP; ++ee) {
        if ((taken >> ee) & 1ull) continue;
        float mv = ((gmask >> (ee >> 3)) & 1u) ? sfc[ee] : 0.0f;
        if (mv > bv) { bv = mv; bi = ee; }
      }
      taken |= 1ull << bi;
      idx[k] = bi;
      wsum += sc[bi];
    }
    float inv = 2.5f / (wsum + 1e-20f);
    for (int k = 0; k < 8; ++k) {
      cw[idx[k]] = sc[idx[k]] * inv;
      topk[t * 8 + k] = idx[k];
      atomicAdd(&counts[idx[k]], 1);
    }
  }
  __syncthreads();
  coef[(size_t)t * NEXP + e] = cw[e];
}

__global__ void moe_zero_counts(int* counts) {
  if (threadIdx.x < NEXP) counts[threadIdx.x] = 0;
}

__global__ void moe_scan(const int* __restrict__ counts, int* __restrict__ offs,
                         int* __restrict__ cursor) {
  if (threadIdx.x == 0) {
    int a = 0;
    for (int e = 0; e < NEXP; ++e) { offs[e] = a; a += counts[e]; }
    offs[NEXP] = a;
  }
  if (threadIdx.x < NEXP) cursor[threadIdx.x] = 0;
}

__global__ void moe_fill(const int* __restrict__ topk, const int* __restrict__ offs,
                         int* __restrict__ cursor, const float* __restrict__ coef,
                         int* __restrict__ rowTok, float* __restrict__ rowCoef) {
  int a = blockIdx.x * blockDim.x + threadIdx.x;
  if (a < T_TOK * 8) {
    int t = a >> 3;
    int e = topk[a];
    int pos = atomicAdd(&cursor[e], 1);
    int row = offs[e] + pos;
    rowTok[row] = t;
    rowCoef[row] = coef[(size_t)t * NEXP + e];
  }
}

// ---------------------------------------------------------------------------
// Shared expert stage 1: Hs[t,i] = bf16( silu(x@SWg) * (x@SWu) )
// Tile 32(M) x 128(N), K-step 64; per wave: 2 N-subtiles x {g,u} = 8 WMMA/stage.
// ---------------------------------------------------------------------------
__global__ __launch_bounds__(256) void moe_shared_gu(
    const float* __restrict__ x, const float* __restrict__ SWg,
    const float* __restrict__ SWu, unsigned short* __restrict__ Hs) {
  __shared__ unsigned int As[32 * 32];
  __shared__ unsigned int Bg[128 * 32];
  __shared__ unsigned int Bu[128 * 32];
  const int tid = threadIdx.x;
  const int lane = tid & 31, wave = tid >> 5;
  const int wm = wave >> 2, wn = wave & 3;
  const int mBase = blockIdx.y * 32;
  const int nBase = blockIdx.x * 128;
  v8f aG0 = {}, aG1 = {}, aU0 = {}, aU1 = {};
  for (int kb = 0; kb < HID; kb += 64) {
    for (int i = tid; i < 32 * 32; i += 256) {
      int r = i >> 5, cp = i & 31;
      const float* g = x + (size_t)(mBase + r) * HID + kb + cp * 2;
      As[r * 32 + cp] = pack2_bf16(g[0], g[1]);
    }
    for (int i = tid; i < 128 * 32; i += 256) {
      int n = i >> 5, kp = i & 31;
      size_t g0 = (size_t)(kb + kp * 2) * ISH + nBase + n;
      Bg[n * 32 + kp] = pack2_bf16(SWg[g0], SWg[g0 + ISH]);
      Bu[n * 32 + kp] = pack2_bf16(SWu[g0], SWu[g0 + ISH]);
    }
    __syncthreads();
    if (kb + 64 < HID) {
      __builtin_prefetch(SWg + (size_t)(kb + 64) * ISH + nBase + (tid & 127), 0, 1);
      __builtin_prefetch(SWu + (size_t)(kb + 64) * ISH + nBase + (tid & 127), 0, 1);
    }
    const unsigned int* arow = As + (wm * 16 + (lane & 15)) * 32;
    const unsigned int* bg0 = Bg + (wn * 16 + (lane & 15)) * 32;
    const unsigned int* bg1 = Bg + (64 + wn * 16 + (lane & 15)) * 32;
    const unsigned int* bu0 = Bu + (wn * 16 + (lane & 15)) * 32;
    const unsigned int* bu1 = Bu + (64 + wn * 16 + (lane & 15)) * 32;
#pragma unroll
    for (int k0 = 0; k0 < 64; k0 += 32) {
      v16bf a = frag_a(arow, k0, lane);
      aG0 = wmma_bf16(a, frag_b(bg0, k0, lane), aG0);
      aG1 = wmma_bf16(a, frag_b(bg1, k0, lane), aG1);
      aU0 = wmma_bf16(a, frag_b(bu0, k0, lane), aU0);
      aU1 = wmma_bf16(a, frag_b(bu1, k0, lane), aU1);
    }
    __syncthreads();
  }
  const int row0 = mBase + wm * 16 + ((lane & 16) ? 8 : 0);
  const int c0 = nBase + wn * 16 + (lane & 15);
#pragma unroll
  for (int v = 0; v < 8; ++v) {
    Hs[(size_t)(row0 + v) * ISH + c0]      = (unsigned short)f32_to_bf16(silu_f(aG0[v]) * aU0[v]);
    Hs[(size_t)(row0 + v) * ISH + c0 + 64] = (unsigned short)f32_to_bf16(silu_f(aG1[v]) * aU1[v]);
  }
}

// Shared expert stage 2: out = Hs @ SWd (initializes d_out).
// A tile staged via global_load_async_to_lds_b128 (bf16 source, pure copy).
__global__ __launch_bounds__(256) void moe_shared_down(
    const unsigned short* __restrict__ Hs, const float* __restrict__ SWd,
    float* __restrict__ out) {
  __shared__ unsigned int As[32 * 32];
  __shared__ unsigned int Bt[128 * 32];
  const int tid = threadIdx.x;
  const int lane = tid & 31, wave = tid >> 5;
  const int wm = wave >> 2, wn = wave & 3;
  const int mBase = blockIdx.y * 32;
  const int nBase = blockIdx.x * 128;
  const unsigned asBase = (unsigned)(size_t)(void*)As;
  const int rowi = tid >> 3, seg = tid & 7;      // 256 thr x 16B = 4KB A tile
  v8f acc0 = {}, acc1 = {};
  for (int kb = 0; kb < ISH; kb += 64) {
    async_g2l_b128(asBase + (unsigned)(rowi * 128 + seg * 16),
                   (const void*)(Hs + (size_t)(mBase + rowi) * ISH + kb + seg * 8));
    for (int i = tid; i < 128 * 32; i += 256) {
      int n = i >> 5, kp = i & 31;
      size_t g0 = (size_t)(kb + kp * 2) * HID + nBase + n;
      Bt[n * 32 + kp] = pack2_bf16(SWd[g0], SWd[g0 + HID]);
    }
    async_wait0();
    __syncthreads();
    if (kb + 64 < ISH)
      __builtin_prefetch(SWd + (size_t)(kb + 64) * HID + nBase + (tid & 127), 0, 1);
    const unsigned int* arow = As + (wm * 16 + (lane & 15)) * 32;
    const unsigned int* b0 = Bt + (wn * 16 + (lane & 15)) * 32;
    const unsigned int* b1 = Bt + (64 + wn * 16 + (lane & 15)) * 32;
#pragma unroll
    for (int k0 = 0; k0 < 64; k0 += 32) {
      v16bf a = frag_a(arow, k0, lane);
      acc0 = wmma_bf16(a, frag_b(b0, k0, lane), acc0);
      acc1 = wmma_bf16(a, frag_b(b1, k0, lane), acc1);
    }
    __syncthreads();
  }
  const int row0 = mBase + wm * 16 + ((lane & 16) ? 8 : 0);
  const int c0 = nBase + wn * 16 + (lane & 15);
#pragma unroll
  for (int v = 0; v < 8; ++v) {
    out[(size_t)(row0 + v) * HID + c0]      = acc0[v];
    out[(size_t)(row0 + v) * HID + c0 + 64] = acc1[v];
  }
}

// ---------------------------------------------------------------------------
// Routed stage 1: Hbuf[row,i] = bf16( silu(x_tok@Wg[e]) * (x_tok@Wu[e]) * coef )
// ---------------------------------------------------------------------------
__global__ __launch_bounds__(256) void moe_routed_gu(
    const float* __restrict__ x, const float* __restrict__ Wg,
    const float* __restrict__ Wu, const int* __restrict__ rowTok,
    const float* __restrict__ rowCoef, const int* __restrict__ counts,
    const int* __restrict__ offs, unsigned short* __restrict__ Hbuf) {
  const int e = blockIdx.z;
  const int cnt = counts[e];
  const int mLoc = blockIdx.y * 32;
  if (mLoc >= cnt) return;
  const int off = offs[e];
  const int nBase = blockIdx.x * 128;
  const float* Wge = Wg + (size_t)e * HID * IEXP;
  const float* Wue = Wu + (size_t)e * HID * IEXP;
  __shared__ unsigned int As[32 * 32];
  __shared__ unsigned int Bg[128 * 32];
  __shared__ unsigned int Bu[128 * 32];
  __shared__ int toks[32];
  const int tid = threadIdx.x;
  if (tid < 32) {
    int r = mLoc + tid;
    toks[tid] = (r < cnt) ? rowTok[off + r] : -1;
  }
  __syncthreads();
  const int lane = tid & 31, wave = tid >> 5;
  const int wm = wave >> 2, wn = wave & 3;
  v8f aG0 = {}, aG1 = {}, aU0 = {}, aU1 = {};
  for (int kb = 0; kb < HID; kb += 64) {
    for (int i = tid; i < 32 * 32; i += 256) {
      int r = i >> 5, cp = i & 31;
      int tok = toks[r];
      int tsafe = (tok >= 0) ? tok : toks[0];        // garbage rows dropped at store
      const float* g = x + (size_t)tsafe * HID + kb + cp * 2;
      As[r * 32 + cp] = pack2_bf16(g[0], g[1]);
    }
    for (int i = tid; i < 128 * 32; i += 256) {
      int n = i >> 5, kp = i & 31;
      size_t g0 = (size_t)(kb + kp * 2) * IEXP + nBase + n;
      Bg[n * 32 + kp] = pack2_bf16(Wge[g0], Wge[g0 + IEXP]);
      Bu[n * 32 + kp] = pack2_bf16(Wue[g0], Wue[g0 + IEXP]);
    }
    __syncthreads();
    if (kb + 64 < HID) {
      __builtin_prefetch(Wge + (size_t)(kb + 64) * IEXP + nBase + (tid & 127), 0, 1);
      __builtin_prefetch(Wue + (size_t)(kb + 64) * IEXP + nBase + (tid & 127), 0, 1);
    }
    const unsigned int* arow = As + (wm * 16 + (lane & 15)) * 32;
    const unsigned int* bg0 = Bg + (wn * 16 + (lane & 15)) * 32;
    const unsigned int* bg1 = Bg + (64 + wn * 16 + (lane & 15)) * 32;
    const unsigned int* bu0 = Bu + (wn * 16 + (lane & 15)) * 32;
    const unsigned int* bu1 = Bu + (64 + wn * 16 + (lane & 15)) * 32;
#pragma unroll
    for (int k0 = 0; k0 < 64; k0 += 32) {
      v16bf a = frag_a(arow, k0, lane);
      aG0 = wmma_bf16(a, frag_b(bg0, k0, lane), aG0);
      aG1 = wmma_bf16(a, frag_b(bg1, k0, lane), aG1);
      aU0 = wmma_bf16(a, frag_b(bu0, k0, lane), aU0);
      aU1 = wmma_bf16(a, frag_b(bu1, k0, lane), aU1);
    }
    __syncthreads();
  }
  const int r0 = wm * 16 + ((lane & 16) ? 8 : 0);
  const int c0 = nBase + wn * 16 + (lane & 15);
#pragma unroll
  for (int v = 0; v < 8; ++v) {
    int r = r0 + v;
    if (mLoc + r < cnt) {
      int rowG = off + mLoc + r;
      float cw = rowCoef[rowG];
      Hbuf[(size_t)rowG * IEXP + c0]      = (unsigned short)f32_to_bf16(silu_f(aG0[v]) * aU0[v] * cw);
      Hbuf[(size_t)rowG * IEXP + c0 + 64] = (unsigned short)f32_to_bf16(silu_f(aG1[v]) * aU1[v] * cw);
    }
  }
}

// Routed stage 2: out[tok,:] += Hbuf_rows @ Wd[e] (f32 atomics).
// A tile staged via async global->LDS (bf16 source).
__global__ __launch_bounds__(256) void moe_routed_down(
    const unsigned short* __restrict__ Hbuf, const float* __restrict__ Wd,
    const int* __restrict__ rowTok, const int* __restrict__ counts,
    const int* __restrict__ offs, float* __restrict__ out) {
  const int e = blockIdx.z;
  const int cnt = counts[e];
  const int mLoc = blockIdx.y * 32;
  if (mLoc >= cnt) return;
  const int off = offs[e];
  const int nBase = blockIdx.x * 128;
  const float* Wde = Wd + (size_t)e * IEXP * HID;
  __shared__ unsigned int As[32 * 32];
  __shared__ unsigned int Bt[128 * 32];
  __shared__ int toks[32];
  const int tid = threadIdx.x;
  if (tid < 32) {
    int r = mLoc + tid;
    toks[tid] = (r < cnt) ? rowTok[off + r] : -1;
  }
  __syncthreads();
  const int lane = tid & 31, wave = tid >> 5;
  const int wm = wave >> 2, wn = wave & 3;
  const unsigned asBase = (unsigned)(size_t)(void*)As;
  const int rowi = tid >> 3, seg = tid & 7;
  const int rsrc = (mLoc + rowi < cnt) ? (mLoc + rowi) : (cnt - 1);  // clamp; dropped rows
  v8f acc0 = {}, acc1 = {};
  for (int kb = 0; kb < IEXP; kb += 64) {
    async_g2l_b128(asBase + (unsigned)(rowi * 128 + seg * 16),
                   (const void*)(Hbuf + (size_t)(off + rsrc) * IEXP + kb + seg * 8));
    for (int i = tid; i < 128 * 32; i += 256) {
      int n = i >> 5, kp = i & 31;
      size_t g0 = (size_t)(kb + kp * 2) * HID + nBase + n;
      Bt[n * 32 + kp] = pack2_bf16(Wde[g0], Wde[g0 + HID]);
    }
    async_wait0();
    __syncthreads();
    if (kb + 64 < IEXP)
      __builtin_prefetch(Wde + (size_t)(kb + 64) * HID + nBase + (tid & 127), 0, 1);
    const unsigned int* arow = As + (wm * 16 + (lane & 15)) * 32;
    const unsigned int* b0 = Bt + (wn * 16 + (lane & 15)) * 32;
    const unsigned int* b1 = Bt + (64 + wn * 16 + (lane & 15)) * 32;
#pragma unroll
    for (int k0 = 0; k0 < 64; k0 += 32) {
      v16bf a = frag_a(arow, k0, lane);
      acc0 = wmma_bf16(a, frag_b(b0, k0, lane), acc0);
      acc1 = wmma_bf16(a, frag_b(b1, k0, lane), acc1);
    }
    __syncthreads();
  }
  const int r0 = wm * 16 + ((lane & 16) ? 8 : 0);
  const int c0 = nBase + wn * 16 + (lane & 15);
#pragma unroll
  for (int v = 0; v < 8; ++v) {
    int tok = toks[r0 + v];
    if (tok >= 0) {
      atomicAdd(&out[(size_t)tok * HID + c0], acc0[v]);
      atomicAdd(&out[(size_t)tok * HID + c0 + 64], acc1[v]);
    }
  }
}

// ---------------------------------------------------------------------------
// Workspace layout (bytes)
// ---------------------------------------------------------------------------
static const size_t OFF_COEF   = 0;                                   // 2048*64*4 = 512KB
static const size_t OFF_COUNTS = 524288;                              // 64 int
static const size_t OFF_CURSOR = OFF_COUNTS + 256;                    // 64 int
static const size_t OFF_OFFS   = OFF_CURSOR + 256;                    // 65 int
static const size_t OFF_TOPK   = OFF_OFFS + 512;                      // 16384 int
static const size_t OFF_ROWTOK = OFF_TOPK + 65536;                    // 16384 int
static const size_t OFF_ROWCO  = OFF_ROWTOK + 65536;                  // 16384 f32
static const size_t OFF_HS     = 1ull << 20;                          // 2048*1024 bf16 = 4MB
static const size_t OFF_HBUF   = OFF_HS + (size_t)T_TOK * ISH * 2;    // 16384*512 bf16 = 16MB

extern "C" void kernel_launch(void* const* d_in, const int* in_sizes, int n_in,
                              void* d_out, int out_size, void* d_ws, size_t ws_size,
                              hipStream_t stream) {
  (void)in_sizes; (void)n_in; (void)out_size; (void)ws_size;
  const float* x   = (const float*)d_in[0];   // [2,1024,1024]
  const float* rw  = (const float*)d_in[1];   // [64,1024]
  const float* eb  = (const float*)d_in[2];   // [64]
  const float* Wg  = (const float*)d_in[3];   // [64,1024,512]
  const float* Wu  = (const float*)d_in[4];   // [64,1024,512]
  const float* Wd  = (const float*)d_in[5];   // [64,512,1024]
  const float* SWg = (const float*)d_in[6];   // [1024,1024]
  const float* SWu = (const float*)d_in[7];   // [1024,1024]
  const float* SWd = (const float*)d_in[8];   // [1024,1024]
  float* out = (float*)d_out;

  char* ws = (char*)d_ws;
  float*          coef    = (float*)(ws + OFF_COEF);
  int*            counts  = (int*)(ws + OFF_COUNTS);
  int*            cursor  = (int*)(ws + OFF_CURSOR);
  int*            offs    = (int*)(ws + OFF_OFFS);
  int*            topk    = (int*)(ws + OFF_TOPK);
  int*            rowTok  = (int*)(ws + OFF_ROWTOK);
  float*          rowCoef = (float*)(ws + OFF_ROWCO);
  unsigned short* Hs      = (unsigned short*)(ws + OFF_HS);
  unsigned short* Hbuf    = (unsigned short*)(ws + OFF_HBUF);

  moe_zero_counts<<<1, 64, 0, stream>>>(counts);
  moe_router<<<T_TOK, 64, 0, stream>>>(x, rw, eb, coef, topk, counts);
  moe_scan<<<1, 64, 0, stream>>>(counts, offs, cursor);
  moe_fill<<<(T_TOK * 8 + 255) / 256, 256, 0, stream>>>(topk, offs, cursor, coef,
                                                        rowTok, rowCoef);
  // shared expert
  moe_shared_gu<<<dim3(ISH / 128, T_TOK / 32), 256, 0, stream>>>(x, SWg, SWu, Hs);
  moe_shared_down<<<dim3(HID / 128, T_TOK / 32), 256, 0, stream>>>(Hs, SWd, out);
  // routed experts (grid covers worst case; blocks past per-expert count exit)
  moe_routed_gu<<<dim3(IEXP / 128, T_TOK / 32, NEXP), 256, 0, stream>>>(
      x, Wg, Wu, rowTok, rowCoef, counts, offs, Hbuf);
  moe_routed_down<<<dim3(HID / 128, T_TOK / 32, NEXP), 256, 0, stream>>>(
      Hbuf, Wd, rowTok, counts, offs, out);
}
